// FullEdgeNodeOnlyModel_23338852286984
// MI455X (gfx1250) — matrix operand order repaired and checked by reading, not compiled, not run on hardware
//
#include <hip/hip_runtime.h>
#include <hip/hip_bf16.h>

typedef _Float16 h8  __attribute__((ext_vector_type(8)));
typedef _Float16 h16 __attribute__((ext_vector_type(16)));
typedef float    f8  __attribute__((ext_vector_type(8)));

#define LDS_STRIDE 72      // staging row stride in halves (144B, 16B-aligned, bank-friendly)
#define N_W 11776          // total padded weight halves in LDS
#define N_B 208            // total padded bias floats in LDS

__device__ __forceinline__ h16 cat2(h8 lo, h8 hi) {
  return __builtin_shufflevector(lo, hi, 0,1,2,3,4,5,6,7,8,9,10,11,12,13,14,15);
}

// A fragment (16x32 f16, M x K): lane holds row m=lane&15; kh=lane>>4 selects K-halves.
// halves [0..7] = K k0..k0+7 (VGPR0-3), [8..15] = K k0+16..k0+23 (VGPR4-7), k0 = kh*8.
__device__ __forceinline__ h16 load_fragA(const _Float16* stag, int m, int kh, int kk) {
  const _Float16* p = stag + m*LDS_STRIDE + kk + kh*8;
  h8 lo = *(const h8*)p;
  h8 hi = *(const h8*)(p + 16);
  return cat2(lo, hi);
}

// B fragment (32x16 f16, K x N): lane holds column n=lane&15; halves [0..7] = K k0..k0+7, [8..15] = K k0+16..k0+23, k0 = kh*8.
// Weights stored transposed Wt[n][k] so this is two contiguous 16B LDS loads.
__device__ __forceinline__ h16 load_fragB(const _Float16* w, int KP, int row, int kh, int kk) {
  const _Float16* p = w + row*KP + kk + kh*8;
  h8 lo = *(const h8*)p;
  h8 hi = *(const h8*)(p + 16);
  return cat2(lo, hi);
}

template<int KITERS, int NTILES>
__device__ __forceinline__ void mlp_layer(_Float16* stag, const _Float16* w, const int KP,
                                          const float* bias, const int lane) {
  const int l15 = lane & 15, kh = lane >> 4;
  h16 a[KITERS];
#pragma unroll
  for (int kt = 0; kt < KITERS; ++kt) a[kt] = load_fragA(stag, l15, kh, kt*32);
  f8 acc[NTILES];
#pragma unroll
  for (int nt = 0; nt < NTILES; ++nt) {
    f8 c = {0.f,0.f,0.f,0.f,0.f,0.f,0.f,0.f};
#pragma unroll
    for (int kt = 0; kt < KITERS; ++kt) {
      h16 b = load_fragB(w, KP, nt*16 + l15, kh, kt*32);
      c = __builtin_amdgcn_wmma_f32_16x16x32_f16(false, a[kt], false, b, (short)0, c, false, false);
    }
    acc[nt] = c;
  }
  __builtin_amdgcn_wave_barrier();
  // D layout: lane n = lane&15; VGPR r holds row m = r + 8*kh. Bias + LeakyReLU, store f16.
#pragma unroll
  for (int nt = 0; nt < NTILES; ++nt) {
    const float bv = bias[nt*16 + l15];
#pragma unroll
    for (int r = 0; r < 8; ++r) {
      float h = acc[nt][r] + bv;
      h = (h >= 0.f) ? h : 0.1f*h;
      stag[(r + 8*kh)*LDS_STRIDE + nt*16 + l15] = (_Float16)h;
    }
  }
}

__global__ __launch_bounds__(256) void edge_mlp_wmma(
    const long long* __restrict__ eidx,
    const float* __restrict__ efeat,
    const _Float16* __restrict__ xn,
    const float* __restrict__ coef,
    const float* __restrict__ W1, const float* __restrict__ W2,
    const float* __restrict__ W3, const float* __restrict__ W4,
    const float* __restrict__ W5, const float* __restrict__ W6,
    const float* __restrict__ B1, const float* __restrict__ B2,
    const float* __restrict__ B3, const float* __restrict__ B4,
    const float* __restrict__ B5, const float* __restrict__ B6,
    float* __restrict__ out, const long long E, const long long nTiles)
{
  __shared__ _Float16 sW[N_W];
  __shared__ float sBias[N_B];
  __shared__ float sES[19], sEB[19];
  __shared__ _Float16 sStag[8*16*LDS_STRIDE];

  const int tid = threadIdx.x;
  {
    const float* Wp[6] = {W1,W2,W3,W4,W5,W6};
    const float* Bp[6] = {B1,B2,B3,B4,B5,B6};
    const int off[6]  = {0,4096,8192,10240,10752,11264};
    const int kp[6]   = {64,64,64,32,32,32};
    const int fin[6]  = {51,64,64,32,16,8};
    const int fout[6] = {64,64,32,16,8,2};
    const int boff[6] = {0,64,128,160,176,192};
    const int nr[6]   = {64,64,32,16,16,16};
    const int wsz[6]  = {4096,4096,2048,512,512,512};
#pragma unroll
    for (int l = 0; l < 6; ++l) {
      for (int i = tid; i < wsz[l]; i += 256) {
        const int n = i / kp[l];
        const int k = i - n*kp[l];
        const float v = (k < fin[l] && n < fout[l]) ? Wp[l][k*fout[l] + n] : 0.f;
        sW[off[l] + i] = (_Float16)v;
      }
      if (tid < nr[l]) sBias[boff[l] + tid] = (tid < fout[l]) ? Bp[l][tid] : 0.f;
    }
  }
  if (tid < 19) { sES[tid] = coef[tid]; sEB[tid] = coef[19 + tid]; }
  __syncthreads();

  const int wave = tid >> 5, lane = tid & 31;
  const int l15 = lane & 15, side = lane >> 4;
  _Float16* stag = sStag + wave*(16*LDS_STRIDE);
  {
    h8 z = {(_Float16)0,(_Float16)0,(_Float16)0,(_Float16)0,
            (_Float16)0,(_Float16)0,(_Float16)0,(_Float16)0};
    for (int i = lane; i < (16*LDS_STRIDE)/8; i += 32) *(h8*)&stag[i*8] = z;
  }
  // Per-lane edge-feature BN coefficients (lanes 0-15: cols 0..15, lanes 16-31: cols 16..18)
  float es[16], eb[16];
#pragma unroll
  for (int c = 0; c < 16; ++c) { es[c] = 0.f; eb[c] = 0.f; }
  if (side == 0) {
#pragma unroll
    for (int c = 0; c < 16; ++c) { es[c] = sES[c]; eb[c] = sEB[c]; }
  } else {
#pragma unroll
    for (int c = 0; c < 3; ++c) { es[c] = sES[16 + c]; eb[c] = sEB[16 + c]; }
  }

  const long long gw = (long long)blockIdx.x*8 + wave;
  const long long gstride = (long long)gridDim.x*8;
  for (long long t = gw; t < nTiles; t += gstride) {
    const long long base = t*16;
    const long long e0 = base + l15;
    const long long ec = (e0 < E) ? e0 : (E - 1);
    // gather: lanes 0-15 fetch src node (cols 0..15), lanes 16-31 fetch dst node (cols 16..31)
    const long long node = eidx[(long long)side*E + ec];
    const _Float16* xp = xn + node*16;
    h8 x0 = *(const h8*)xp;
    h8 x1 = *(const h8*)(xp + 8);
    *(h8*)&stag[l15*LDS_STRIDE + side*16]     = x0;
    *(h8*)&stag[l15*LDS_STRIDE + side*16 + 8] = x1;
    // edge features: BN applied on the fly, f32 -> f16 into cols 32..50
    const float* ep = efeat + ec*19;
    if (side == 0) {
#pragma unroll
      for (int c = 0; c < 16; ++c)
        stag[l15*LDS_STRIDE + 32 + c] = (_Float16)(ep[c]*es[c] + eb[c]);
    } else {
#pragma unroll
      for (int c = 0; c < 3; ++c)
        stag[l15*LDS_STRIDE + 48 + c] = (_Float16)(ep[16 + c]*es[c] + eb[c]);
    }
    __builtin_amdgcn_wave_barrier();
    mlp_layer<2,4>(stag, sW + 0,     64, sBias + 0,   lane);  // 51(pad 64)->64
    __builtin_amdgcn_wave_barrier();
    mlp_layer<2,4>(stag, sW + 4096,  64, sBias + 64,  lane);  // 64->64
    __builtin_amdgcn_wave_barrier();
    mlp_layer<2,2>(stag, sW + 8192,  64, sBias + 128, lane);  // 64->32
    __builtin_amdgcn_wave_barrier();
    mlp_layer<1,1>(stag, sW + 10240, 32, sBias + 160, lane);  // 32->16
    __builtin_amdgcn_wave_barrier();
    mlp_layer<1,1>(stag, sW + 10752, 32, sBias + 176, lane);  // 16(pad 32)->8(pad 16)
    __builtin_amdgcn_wave_barrier();
    {
      // last layer: 8(pad 32)->2(pad 16), no LeakyReLU, write fp32 to global
      h16 a = load_fragA(stag, l15, side, 0);
      h16 b = load_fragB(sW + 11264, 32, l15, side, 0);
      f8 c = {0.f,0.f,0.f,0.f,0.f,0.f,0.f,0.f};
      c = __builtin_amdgcn_wmma_f32_16x16x32_f16(false, a, false, b, (short)0, c, false, false);
      if (l15 < 2) {
        const float bv = sBias[192 + l15];
#pragma unroll
        for (int r = 0; r < 8; ++r) {
          const long long row = base + r + 8*side;
          if (row < E) out[row*2 + l15] = c[r] + bv;
        }
      }
    }
    __builtin_amdgcn_wave_barrier();
  }
}

// ---- deterministic two-stage BN statistics ----
__global__ __launch_bounds__(256) void bn_partial(const float* __restrict__ src, const long long N,
                                                  const int F, const int fbase,
                                                  float* __restrict__ psum, float* __restrict__ psq)
{
  const int f = blockIdx.y;
  const int chunk = blockIdx.x;              // fixed 256 chunks -> deterministic
  const long long rpc = (N + 255) / 256;
  const long long start = (long long)chunk * rpc;
  const long long end = (start + rpc < N) ? (start + rpc) : N;
  float s = 0.f, q = 0.f;
  for (long long i = start + threadIdx.x; i < end; i += 256) {
    const float v = src[i*(long long)F + f];
    s += v; q += v*v;
  }
  __shared__ float ls[256], lq[256];
  ls[threadIdx.x] = s; lq[threadIdx.x] = q;
  __syncthreads();
  for (int w = 128; w > 0; w >>= 1) {
    if (threadIdx.x < w) { ls[threadIdx.x] += ls[threadIdx.x + w]; lq[threadIdx.x] += lq[threadIdx.x + w]; }
    __syncthreads();
  }
  if (threadIdx.x == 0) {
    psum[(fbase + f)*256 + chunk] = ls[0];
    psq [(fbase + f)*256 + chunk] = lq[0];
  }
}

__global__ __launch_bounds__(256) void bn_finalize(const float* __restrict__ psum, const float* __restrict__ psq,
                                                   const float* __restrict__ gE, const float* __restrict__ bE,
                                                   const float* __restrict__ gX, const float* __restrict__ bX,
                                                   const long long NE, const long long NX,
                                                   float* __restrict__ coef)
{
  const int b = blockIdx.x;                  // 0..18 = edge features, 19..34 = node features
  const bool isE = (b < 19);
  const int f = isE ? b : (b - 19);
  __shared__ float ls[256], lq[256];
  ls[threadIdx.x] = psum[b*256 + threadIdx.x];
  lq[threadIdx.x] = psq [b*256 + threadIdx.x];
  __syncthreads();
  for (int w = 128; w > 0; w >>= 1) {
    if (threadIdx.x < w) { ls[threadIdx.x] += ls[threadIdx.x + w]; lq[threadIdx.x] += lq[threadIdx.x + w]; }
    __syncthreads();
  }
  if (threadIdx.x == 0) {
    const double n = (double)(isE ? NE : NX);
    const float mean = (float)((double)ls[0] / n);
    float var = (float)((double)lq[0] / n) - mean*mean;
    var = var > 0.f ? var : 0.f;
    const float g  = isE ? gE[f] : gX[f];
    const float bb = isE ? bE[f] : bX[f];
    const float scale = g / sqrtf(var + 1e-5f);
    const float shift = bb - mean*scale;
    if (isE) { coef[f]      = scale; coef[19 + f] = shift; }
    else     { coef[38 + f] = scale; coef[54 + f] = shift; }
  }
}

__global__ __launch_bounds__(256) void bn_make_xn(const float* __restrict__ x, const float* __restrict__ coef,
                                                  _Float16* __restrict__ xn, const long long total)
{
  const long long i = (long long)blockIdx.x*256 + threadIdx.x;
  if (i < total) {
    const int c = (int)(i & 15);
    xn[i] = (_Float16)(x[i]*coef[38 + c] + coef[54 + c]);
  }
}

extern "C" void kernel_launch(void* const* d_in, const int* in_sizes, int n_in,
                              void* d_out, int out_size, void* d_ws, size_t ws_size,
                              hipStream_t stream)
{
  (void)n_in; (void)out_size; (void)ws_size;
  const float*     x    = (const float*)d_in[0];
  const float*     e    = (const float*)d_in[1];
  const long long* eidx = (const long long*)d_in[2];
  const float*     gX   = (const float*)d_in[4];
  const float*     bX   = (const float*)d_in[5];
  const float*     gE   = (const float*)d_in[6];
  const float*     bE   = (const float*)d_in[7];
  const float* W[6]; const float* B[6];
  for (int i = 0; i < 6; ++i) { W[i] = (const float*)d_in[8 + 2*i]; B[i] = (const float*)d_in[9 + 2*i]; }

  const long long NX = in_sizes[0] / 16;
  const long long NE = in_sizes[1] / 19;

  // workspace layout: psum[35*256] | psq[35*256] | coef[70] | xn (f16, NX*16)
  char*     ws   = (char*)d_ws;
  float*    psum = (float*)ws;
  float*    psq  = psum + 35*256;
  float*    coef = psq  + 35*256;
  _Float16* xn   = (_Float16*)(ws + 72064);

  bn_partial<<<dim3(256,19), 256, 0, stream>>>(e, NE, 19, 0,  psum, psq);
  bn_partial<<<dim3(256,16), 256, 0, stream>>>(x, NX, 16, 19, psum, psq);
  bn_finalize<<<35, 256, 0, stream>>>(psum, psq, gE, bE, gX, bX, NE, NX, coef);

  const long long totX = NX*16;
  bn_make_xn<<<(unsigned)((totX + 255)/256), 256, 0, stream>>>(x, coef, xn, totX);

  const long long nTiles = (NE + 15)/16;
  edge_mlp_wmma<<<1024, 256, 0, stream>>>(eidx, e, xn, coef,
      W[0],W[1],W[2],W[3],W[4],W[5],
      B[0],B[1],B[2],B[3],B[4],B[5],
      (float*)d_out, NE, nTiles);
}
